// RJSPGNN_47124381172096
// MI455X (gfx1250) — matrix operand order repaired and checked by reading, not compiled, not run on hardware
//
#include <hip/hip_runtime.h>

#define H 128

typedef __attribute__((ext_vector_type(16))) __bf16 v16bf;
typedef __attribute__((ext_vector_type(8)))  float  v8f;

__device__ __forceinline__ unsigned short f2bf(float f) {
    unsigned int u = __float_as_uint(f);
    u += 0x7FFFu + ((u >> 16) & 1u);       // round-to-nearest-even
    return (unsigned short)(u >> 16);
}

// ---------------- encoders: h = x @ Wenc + b  (fan_in = 3 or 4) ----------------
__global__ void __launch_bounds__(256)
encode3(const float* __restrict__ x, const float* __restrict__ Wenc,
        const float* __restrict__ b, float* __restrict__ h, int N) {
    int t = blockIdx.x * 256 + threadIdx.x;
    if (t >= N * H) return;
    int i = t >> 7, hh = t & (H - 1);
    h[t] = b[hh] + x[i*3+0]*Wenc[0*H+hh] + x[i*3+1]*Wenc[1*H+hh] + x[i*3+2]*Wenc[2*H+hh];
}

__global__ void __launch_bounds__(256)
encode4(const float* __restrict__ x, const float* __restrict__ Wenc,
        const float* __restrict__ b, float* __restrict__ h, int N) {
    int t = blockIdx.x * 256 + threadIdx.x;
    if (t >= N * H) return;
    int i = t >> 7, hh = t & (H - 1);
    h[t] = b[hh] + x[i*4+0]*Wenc[0*H+hh] + x[i*4+1]*Wenc[1*H+hh]
                 + x[i*4+2]*Wenc[2*H+hh] + x[i*4+3]*Wenc[3*H+hh];
}

// ---------------- utility ----------------
__global__ void __launch_bounds__(256)
zero_f32(float* __restrict__ p, int n) {
    int t = blockIdx.x * 256 + threadIdx.x;
    if (t < n) p[t] = 0.0f;
}

__global__ void __launch_bounds__(256)
add_bf16(const float* __restrict__ a, const float* __restrict__ b,
         unsigned short* __restrict__ o, int n) {
    int t = blockIdx.x * 256 + threadIdx.x;
    if (t < n) o[t] = f2bf(a[t] + b[t]);
}

__global__ void __launch_bounds__(256)
mean_bf16(const float* __restrict__ a, const float* __restrict__ cnt,
          unsigned short* __restrict__ o, int n) {
    int t = blockIdx.x * 256 + threadIdx.x;
    if (t < n) o[t] = f2bf(a[t] / fmaxf(cnt[t >> 7], 1.0f));
}

__global__ void __launch_bounds__(256)
to_bf16(const float* __restrict__ a, unsigned short* __restrict__ o, int n) {
    int t = blockIdx.x * 256 + threadIdx.x;
    if (t < n) o[t] = f2bf(a[t]);
}

// ---------------- GINE edge messages: relu(h_src[j] + ea@We + be) scatter-add ----------------
__global__ void __launch_bounds__(256)
gine_edge3(const float* __restrict__ hsrc, const int* __restrict__ ei,
           const float* __restrict__ ea, const float* __restrict__ We,
           const float* __restrict__ be, float* __restrict__ acc, int E) {
    long long t = (long long)blockIdx.x * 256 + threadIdx.x;
    if (t >= (long long)E * H) return;
    int e = (int)(t >> 7), hh = (int)t & (H - 1);
    int src = ei[e], dst = ei[E + e];
    float m = hsrc[(size_t)src * H + hh] + be[hh]
            + ea[(size_t)e*3+0] * We[0*H+hh]
            + ea[(size_t)e*3+1] * We[1*H+hh]
            + ea[(size_t)e*3+2] * We[2*H+hh];
    m = fmaxf(m, 0.0f);
    atomicAdd(&acc[(size_t)dst * H + hh], m);
}

__global__ void __launch_bounds__(256)
gine_edge1(const float* __restrict__ hsrc, const int* __restrict__ ei,
           const float* __restrict__ ea, const float* __restrict__ We,
           const float* __restrict__ be, float* __restrict__ acc, int E) {
    long long t = (long long)blockIdx.x * 256 + threadIdx.x;
    if (t >= (long long)E * H) return;
    int e = (int)(t >> 7), hh = (int)t & (H - 1);
    int src = ei[e], dst = ei[E + e];
    float m = hsrc[(size_t)src * H + hh] + be[hh] + ea[e] * We[hh];
    m = fmaxf(m, 0.0f);
    atomicAdd(&acc[(size_t)dst * H + hh], m);
}

// ---------------- SAGE scatter (sum) + degree count ----------------
__global__ void __launch_bounds__(256)
sage_scatter(const float* __restrict__ hsrc, const int* __restrict__ ei,
             float* __restrict__ acc, int E) {
    long long t = (long long)blockIdx.x * 256 + threadIdx.x;
    if (t >= (long long)E * H) return;
    int e = (int)(t >> 7), hh = (int)t & (H - 1);
    int src = ei[e], dst = ei[E + e];
    atomicAdd(&acc[(size_t)dst * H + hh], hsrc[(size_t)src * H + hh]);
}

__global__ void __launch_bounds__(256)
sage_count(const int* __restrict__ ei, float* __restrict__ cnt, int E) {
    int e = blockIdx.x * 256 + threadIdx.x;
    if (e < E) atomicAdd(&cnt[ei[E + e]], 1.0f);
}

// ---------------- WMMA GEMM: C[M,128] (+)= A_bf16[M,128] @ W[128,128] (+ bias) ----------------
// 256 threads = 8 waves; block computes a 64-row x 128-col stripe (4 row-tiles per wave).
// W is pre-swizzled in LDS into the exact per-lane B-fragment layout, split in two
// 8-element planes so each fragment load is two ds_load_b128 (no scalar u16 gathers).
__global__ void __launch_bounds__(256)
gemm_bf16_128(const unsigned short* __restrict__ A, const float* __restrict__ W,
              const float* __restrict__ bias, float* __restrict__ C,
              int M, int accumulate) {
    // plane p (0/1), colTile w (0..7), kStep ks (0..3), lane (0..31), elem i8 (0..7)
    // value = bf16( W[ ks*32 + p*16 + (lane<16?0:8) + i8 ][ w*16 + (lane&15) ] )
    __shared__ unsigned short Wlds[2 * 8 * 4 * 32 * 8];   // 32 KB
    for (int idx = threadIdx.x; idx < H * H; idx += 256) {
        int i    = idx & 15;          // 0..15 fragment element
        int lane = (idx >> 4) & 31;
        int ks   = (idx >> 9) & 3;
        int w    = idx >> 11;
        int col  = w * 16 + (lane & 15);
        int k    = ks * 32 + (i >> 3) * 16 + ((lane < 16) ? 0 : 8) + (i & 7);
        int plane = i >> 3;
        int pos   = plane * 8192 + ((w * 4 + ks) * 32 + lane) * 8 + (i & 7);
        Wlds[pos] = f2bf(W[k * H + col]);
    }
    __syncthreads();

    const int wave = threadIdx.x >> 5;
    const int lane = threadIdx.x & 31;
    const int half = lane >> 4;                       // K-half select
    const int l16  = lane & 15;
    const int col  = wave * 16 + l16;
    const int row0 = blockIdx.x * 64;

    // A fragment rows for the 4 row-tiles (clamped for the tail block)
    int arow[4];
    #pragma unroll
    for (int rt = 0; rt < 4; ++rt) {
        int r = row0 + rt * 16 + l16;
        arow[rt] = (r > M - 1) ? (M - 1) : r;
    }

    v8f c0 = {}, c1 = {}, c2 = {}, c3 = {};
    #pragma unroll
    for (int ks = 0; ks < 4; ++ks) {
        const int k0 = ks * 32;
        // B 32x16 fragment: two contiguous 128-bit LDS loads
        union { v16bf v; uint4 q[2]; } b;
        const unsigned short* bp = Wlds + ((wave * 4 + ks) * 32 + lane) * 8;
        b.q[0] = *reinterpret_cast<const uint4*>(bp);
        b.q[1] = *reinterpret_cast<const uint4*>(bp + 8192);

        // 4 row-tiles share this B fragment
        union { v16bf v; uint4 q[2]; } a;
        const unsigned short* ap;

        ap = A + (size_t)arow[0] * H + k0 + half * 8;
        a.q[0] = *reinterpret_cast<const uint4*>(ap);
        a.q[1] = *reinterpret_cast<const uint4*>(ap + 16);
        c0 = __builtin_amdgcn_wmma_f32_16x16x32_bf16(false, a.v, false, b.v, (short)0, c0, false, false);

        ap = A + (size_t)arow[1] * H + k0 + half * 8;
        a.q[0] = *reinterpret_cast<const uint4*>(ap);
        a.q[1] = *reinterpret_cast<const uint4*>(ap + 16);
        c1 = __builtin_amdgcn_wmma_f32_16x16x32_bf16(false, a.v, false, b.v, (short)0, c1, false, false);

        ap = A + (size_t)arow[2] * H + k0 + half * 8;
        a.q[0] = *reinterpret_cast<const uint4*>(ap);
        a.q[1] = *reinterpret_cast<const uint4*>(ap + 16);
        c2 = __builtin_amdgcn_wmma_f32_16x16x32_bf16(false, a.v, false, b.v, (short)0, c2, false, false);

        ap = A + (size_t)arow[3] * H + k0 + half * 8;
        a.q[0] = *reinterpret_cast<const uint4*>(ap);
        a.q[1] = *reinterpret_cast<const uint4*>(ap + 16);
        c3 = __builtin_amdgcn_wmma_f32_16x16x32_bf16(false, a.v, false, b.v, (short)0, c3, false, false);
    }

    const float bv = bias ? bias[col] : 0.0f;
    v8f acc[4] = {c0, c1, c2, c3};
    #pragma unroll
    for (int rt = 0; rt < 4; ++rt) {
        #pragma unroll
        for (int i = 0; i < 8; ++i) {
            int m = row0 + rt * 16 + half * 8 + i;
            if (m < M) {
                size_t idx = (size_t)m * H + col;
                float v = acc[rt][i] + bv;
                if (accumulate) v += C[idx];
                C[idx] = v;
            }
        }
    }
}

// ---------------- final column-mean reduction into d_out[256] ----------------
__global__ void __launch_bounds__(256)
final_reduce(const float* __restrict__ out_m, const float* __restrict__ out_o,
             float* __restrict__ out, int NMn, int NOn) {
    __shared__ float red[256];
    const int b = blockIdx.x, t = threadIdx.x;
    const float* src; int n, col; float scale;
    if (b < H) { src = out_m; n = NMn; col = b;     scale = 1.0f / (float)NMn; }
    else       { src = out_o; n = NOn; col = b - H; scale = 1.0f / (3.0f * (float)NOn); }
    float s = 0.0f;
    for (int r = t; r < n; r += 256) s += src[(size_t)r * H + col];
    red[t] = s; __syncthreads();
    for (int k = 128; k > 0; k >>= 1) {
        if (t < k) red[t] += red[t + k];
        __syncthreads();
    }
    if (t == 0) out[b] = red[0] * scale;
}

extern "C" void kernel_launch(void* const* d_in, const int* in_sizes, int n_in,
                              void* d_out, int out_size, void* d_ws, size_t ws_size,
                              hipStream_t stream) {
    const float* x_m    = (const float*)d_in[0];
    const float* x_o    = (const float*)d_in[1];
    const float* ea_a   = (const float*)d_in[2];
    const float* ea_c   = (const float*)d_in[3];
    const int*   ei_a   = (const int*)d_in[4];
    const int*   ei_c   = (const int*)d_in[5];
    const int*   ei_tv  = (const int*)d_in[6];
    const int*   ei_log = (const int*)d_in[7];
    const float* Wm   = (const float*)d_in[8];  const float* bm   = (const float*)d_in[9];
    const float* Wo   = (const float*)d_in[10]; const float* bo   = (const float*)d_in[11];
    const float* Wn_a = (const float*)d_in[12]; const float* bn_a = (const float*)d_in[13];
    const float* We_a = (const float*)d_in[14]; const float* be_a = (const float*)d_in[15];
    const float* Wn_c = (const float*)d_in[16]; const float* bn_c = (const float*)d_in[17];
    const float* We_c = (const float*)d_in[18]; const float* be_c = (const float*)d_in[19];
    const float* Wl_tv  = (const float*)d_in[20]; const float* bl_tv  = (const float*)d_in[21];
    const float* Wr_tv  = (const float*)d_in[22];
    const float* Wl_log = (const float*)d_in[23]; const float* bl_log = (const float*)d_in[24];
    const float* Wr_log = (const float*)d_in[25];

    const int NM = in_sizes[0] / 3;
    const int NO = in_sizes[1] / 4;
    const int EA = in_sizes[4] / 2;
    const int EC = in_sizes[5] / 2;
    const int ET = in_sizes[6] / 2;
    const int EL = in_sizes[7] / 2;

    char* ws = (char*)d_ws;
    auto alloc = [&](size_t bytes) {
        char* p = ws; ws += (bytes + 255) & ~(size_t)255; return p;
    };
    float* h_o   = (float*)alloc((size_t)NO * H * 4);
    float* h_m   = (float*)alloc((size_t)NM * H * 4);
    float* acc   = (float*)alloc((size_t)NO * H * 4);   // scatter accumulator (L2-resident)
    float* out_o = (float*)alloc((size_t)NO * H * 4);   // o_a + o_c + o_log
    float* out_m = (float*)alloc((size_t)NM * H * 4);   // m_tv
    float* cnt   = (float*)alloc((size_t)NO * 4);
    unsigned short* Abf  = (unsigned short*)alloc((size_t)NO * H * 2);
    unsigned short* hobf = (unsigned short*)alloc((size_t)NO * H * 2);
    unsigned short* hmbf = (unsigned short*)alloc((size_t)NM * H * 2);

    auto nb = [](long long n) { return (unsigned)((n + 255) / 256); };
    auto gb = [](int m) { return (unsigned)((m + 63) / 64); };   // 64-row GEMM stripes

    // encoders
    encode3<<<nb((long long)NM * H), 256, 0, stream>>>(x_m, Wm, bm, h_m, NM);
    encode4<<<nb((long long)NO * H), 256, 0, stream>>>(x_o, Wo, bo, h_o, NO);

    // GINE (assignment): machine -> operation
    zero_f32<<<nb((long long)NO * H), 256, 0, stream>>>(acc, NO * H);
    gine_edge3<<<nb((long long)EA * H), 256, 0, stream>>>(h_m, ei_a, ea_a, We_a, be_a, acc, EA);
    add_bf16<<<nb((long long)NO * H), 256, 0, stream>>>(acc, h_o, Abf, NO * H);
    gemm_bf16_128<<<gb(NO), 256, 0, stream>>>(Abf, Wn_a, bn_a, out_o, NO, 0);

    // GINE (completion): operation -> operation
    zero_f32<<<nb((long long)NO * H), 256, 0, stream>>>(acc, NO * H);
    gine_edge1<<<nb((long long)EC * H), 256, 0, stream>>>(h_o, ei_c, ea_c, We_c, be_c, acc, EC);
    add_bf16<<<nb((long long)NO * H), 256, 0, stream>>>(acc, h_o, Abf, NO * H);
    gemm_bf16_128<<<gb(NO), 256, 0, stream>>>(Abf, Wn_c, bn_c, out_o, NO, 1);

    // SAGE (logical): operation -> operation, mean aggregation
    zero_f32<<<nb((long long)NO * H), 256, 0, stream>>>(acc, NO * H);
    zero_f32<<<nb(NO), 256, 0, stream>>>(cnt, NO);
    sage_scatter<<<nb((long long)EL * H), 256, 0, stream>>>(h_o, ei_log, acc, EL);
    sage_count<<<nb(EL), 256, 0, stream>>>(ei_log, cnt, EL);
    mean_bf16<<<nb((long long)NO * H), 256, 0, stream>>>(acc, cnt, Abf, NO * H);
    gemm_bf16_128<<<gb(NO), 256, 0, stream>>>(Abf, Wl_log, bl_log, out_o, NO, 1);
    to_bf16<<<nb((long long)NO * H), 256, 0, stream>>>(h_o, hobf, NO * H);
    gemm_bf16_128<<<gb(NO), 256, 0, stream>>>(hobf, Wr_log, nullptr, out_o, NO, 1);

    // SAGE (type_valid): operation -> machine, mean aggregation
    zero_f32<<<nb((long long)NM * H), 256, 0, stream>>>(acc, NM * H);
    zero_f32<<<nb(NM), 256, 0, stream>>>(cnt, NM);
    sage_scatter<<<nb((long long)ET * H), 256, 0, stream>>>(h_o, ei_tv, acc, ET);
    sage_count<<<nb(ET), 256, 0, stream>>>(ei_tv, cnt, ET);
    mean_bf16<<<nb((long long)NM * H), 256, 0, stream>>>(acc, cnt, Abf, NM * H);
    gemm_bf16_128<<<gb(NM), 256, 0, stream>>>(Abf, Wl_tv, bl_tv, out_m, NM, 0);
    to_bf16<<<nb((long long)NM * H), 256, 0, stream>>>(h_m, hmbf, NM * H);
    gemm_bf16_128<<<gb(NM), 256, 0, stream>>>(hmbf, Wr_tv, nullptr, out_m, NM, 1);

    // global mean pooling -> [1, 2H]
    final_reduce<<<2 * H, 256, 0, stream>>>(out_m, out_o, (float*)d_out, NM, NO);
}
